// SpatialNet_26448408609392
// MI455X (gfx1250) — compile-verified, matrix operand order checked by reading, not run on hardware
//
#include <hip/hip_runtime.h>

// ---------------- SGN fused for gfx1250 (MI455X), wave32 + WMMA bf16 -------
// One workgroup (256 threads = 8 waves) handles one (batch, frame) pair.
// All activations stay in LDS in WMMA-operand-swizzled layout: B-operand
// loads are 2x ds_load_b128, accumulator stores are 1x ds_store_b128, and
// all weights (incl. the embed 64x64 layers) are pre-swizzled in global so
// A-operands are 2x global_load_b128.  Every GEMM in the net (embed layer-2,
// q/k projections, attention scores, graph aggregation, gcn convs) runs on
// v_wmma_f32_16x16x32_bf16.

#define J   25
#define JP  32          // joint dim padded to 2 WMMA N-tiles
#define TT  256
#define BB  32
#define RS  0.9999950000374997f   // (1+1e-5)^-0.5

typedef __bf16 bf16_t;
typedef bf16_t v16bf __attribute__((ext_vector_type(16)));
typedef bf16_t v8bf  __attribute__((ext_vector_type(8)));
typedef float  v8f   __attribute__((ext_vector_type(8)));

// K offset inside a 16-bit A/B WMMA operand tile (ISA 7.12.2, 16-bit 16x32)
__device__ __host__ __forceinline__ int koff(int e, int half) {
    int vg = e >> 1, pr = e & 1;
    return ((vg & 4) << 2) + (half << 3) + ((vg & 3) << 1) + pr;
}

// Element index of value (k,n) inside a swizzled K x 32 matrix
// (tiles of 32K x 16N, 512 elems each, tile order (k>>5)*2 + (n>>4);
//  within a tile, lane*16 + e with e contiguous per (lane, k-octet)).
__device__ __forceinline__ int swz_idx(int k, int n) {
    int k5 = k & 31;
    int tile = ((k >> 5) << 1) + (n >> 4);
    int lane_b = (n & 15) + (((k5 >> 3) & 1) << 4);
    int e = ((k5 >> 4) << 3) + (k5 & 7);
    return (tile << 9) + (lane_b << 4) + e;
}

__device__ __forceinline__ v16bf join16(v8bf lo, v8bf hi) {
    union { v16bf v; v8bf h[2]; } u;
    u.h[0] = lo; u.h[1] = hi;
    return u.v;
}

// B operand (K x 16 tile at (k0,n0)) from swizzled storage: 2x b128 loads
__device__ __forceinline__ v16bf load_b_swz(const bf16_t* base, int k0, int n0,
                                            int lane) {
    const bf16_t* p = base + ((((k0 >> 5) << 1) + (n0 >> 4)) << 9) + (lane << 4);
    return join16(*(const v8bf*)p, *(const v8bf*)(p + 8));
}

// A operand from pre-swizzled weights in global memory: 2x b128 loads
__device__ __forceinline__ v16bf load_a_w(const bf16_t* base, int ktiles,
                                          int mt, int kt, int lane) {
    const bf16_t* p = base + ((((mt * ktiles + kt) << 5) + lane) << 4);
    return join16(*(const v8bf*)p, *(const v8bf*)(p + 8));
}

// A operand where the matrix lives in swizzled (K=row, N=col) activation
// storage: A[m][k] = act(row=m, col=k).  (minor path: aggregation x)
__device__ __forceinline__ v16bf load_a_act(const bf16_t* base, int m0, int k0,
                                            int lane) {
    int half = lane >> 4;
    int m = m0 + (lane & 15);
    v16bf a;
#pragma unroll
    for (int e = 0; e < 16; ++e)
        a[e] = base[swz_idx(m, k0 + koff(e, half))];
    return a;
}

// A operand = transpose of swizzled activation: A[m][k] = act(row=k, col=m).
// (minor path: q^T in the attention scores)
__device__ __forceinline__ v16bf load_at_act(const bf16_t* base, int m0, int k0,
                                             int lane) {
    int half = lane >> 4;
    int m = m0 + (lane & 15);
    v16bf a;
#pragma unroll
    for (int e = 0; e < 16; ++e)
        a[e] = base[swz_idx(k0 + koff(e, half), m)];
    return a;
}

__device__ __forceinline__ v8f wmma_bf16(v16bf a, v16bf b, v8f c) {
    // (neg_a, A, neg_b, B, c_mod, C, reuse_a, reuse_b)
    return __builtin_amdgcn_wmma_f32_16x16x32_bf16(false, a, false, b,
                                                   (short)0, c, false, false);
}

// Store an f32 accumulator tile into swizzled bf16 storage: the 8 rows a lane
// owns are contiguous in the swizzled layout -> ONE ds_store_b128.
__device__ __forceinline__ void store_tile_swz(bf16_t* dst, int m0, int n0,
                                               v8f acc, int lane) {
    int half = lane >> 4, col = lane & 15;
    int base = swz_idx(m0 + half * 8, n0 + col);
    v8bf pk;
#pragma unroll
    for (int r = 0; r < 8; ++r) pk[r] = (bf16_t)acc[r];
    *(v8bf*)(dst + base) = pk;
}

// Same but with BatchNorm + ReLU epilogue (per-row scale/shift)
__device__ __forceinline__ void store_tile_bn(bf16_t* dst, int m0, int n0,
                                              v8f acc, int lane,
                                              const float* __restrict__ bng,
                                              const float* __restrict__ bnb) {
    int half = lane >> 4, col = lane & 15;
    int mb = m0 + half * 8;
    int base = swz_idx(mb, n0 + col);
    v8bf pk;
#pragma unroll
    for (int r = 0; r < 8; ++r) {
        float v = acc[r] * (bng[mb + r] * RS) + bnb[mb + r];
        pk[r] = (bf16_t)fmaxf(v, 0.f);
    }
    *(v8bf*)(dst + base) = pk;
}

// prologue: fp32 weight [O][C] -> bf16 in A-operand-swizzled tile order
__global__ void swz_weight(const float* __restrict__ src,
                           bf16_t* __restrict__ dst, int O, int C) {
    int i = blockIdx.x * 256 + threadIdx.x;
    if (i >= O * C) return;
    int tile = i >> 9, r = i & 511;
    int lane = r >> 4, e = r & 15;
    int ktiles = C >> 5;
    int mt = tile / ktiles, kt = tile - mt * ktiles;
    int m = (mt << 4) + (lane & 15);
    int k = (kt << 5) + koff(e, lane >> 4);
    dst[i] = (bf16_t)src[m * C + k];
}

// one gcn_spa layer: agg = x * gT (WMMA), y = Wa*agg + Wb*x + bias, BN, relu
__device__ void gcn_layer(int C, int O, int lane, int wave,
                          const bf16_t* xS, bf16_t* aggS, bf16_t* outS,
                          const bf16_t* gTS,
                          const bf16_t* __restrict__ Wa,
                          const bf16_t* __restrict__ Wb,
                          const float* __restrict__ bias,
                          const float* __restrict__ bng,
                          const float* __restrict__ bnb) {
    int half = lane >> 4;
    // --- aggregation: M=C, N=JP, K=JP (one WMMA k-step) ---
    int njobs = (C >> 4) * 2;
    for (int job = wave; job < njobs; job += 8) {
        int m0 = (job >> 1) << 4, n0 = (job & 1) << 4;
        v16bf a = load_a_act(xS, m0, 0, lane);
        v16bf bm = load_b_swz(gTS, 0, n0, lane);
        v8f acc = {};
        acc = wmma_bf16(a, bm, acc);
        store_tile_swz(aggS, m0, n0, acc, lane);
    }
    __syncthreads();
    // --- conv pair + BN + relu: M=O, N=JP, K=C ---
    int njobs2 = (O >> 4) * 2;
    int ksteps = C >> 5;
    for (int job = wave; job < njobs2; job += 8) {
        int m0 = (job >> 1) << 4, n0 = (job & 1) << 4;
        int mt = m0 >> 4;
        v8f acc;
#pragma unroll
        for (int r = 0; r < 8; ++r) acc[r] = bias[m0 + r + half * 8];
        for (int kk = 0; kk < ksteps; ++kk) {
            v16bf a1 = load_a_w(Wa, ksteps, mt, kk, lane);
            v16bf b1 = load_b_swz(aggS, kk * 32, n0, lane);
            acc = wmma_bf16(a1, b1, acc);
            v16bf a2 = load_a_w(Wb, ksteps, mt, kk, lane);
            v16bf b2 = load_b_swz(xS, kk * 32, n0, lane);
            acc = wmma_bf16(a2, b2, acc);
        }
        store_tile_bn(outS, m0, n0, acc, lane, bng, bnb);
    }
    __syncthreads();
}

#define POOL_BYTES 51200

__global__ __launch_bounds__(256) void sgn_fused(
    const float* __restrict__ x, const float* __restrict__ spa,
    const float* __restrict__ jbn_g, const float* __restrict__ jbn_b,
    const float* __restrict__ jw1, const float* __restrict__ jb1,
    const float* __restrict__ jb2,
    const float* __restrict__ dbn_g, const float* __restrict__ dbn_b,
    const float* __restrict__ dw1, const float* __restrict__ db1,
    const float* __restrict__ db2,
    const float* __restrict__ sw1, const float* __restrict__ sb1,
    const float* __restrict__ sb2,
    const float* __restrict__ g1b, const float* __restrict__ g2b,
    const float* __restrict__ w1bb, const float* __restrict__ bn1g,
    const float* __restrict__ bn1b, const float* __restrict__ w2bb,
    const float* __restrict__ bn2g, const float* __restrict__ bn2b,
    const float* __restrict__ w3bb, const float* __restrict__ bn3g,
    const float* __restrict__ bn3b,
    const bf16_t* __restrict__ jw2B, const bf16_t* __restrict__ dw2B,
    const bf16_t* __restrict__ sw2B,
    const bf16_t* __restrict__ g1wB, const bf16_t* __restrict__ g2wB,
    const bf16_t* __restrict__ w1aB, const bf16_t* __restrict__ w1bB,
    const bf16_t* __restrict__ w2aB, const bf16_t* __restrict__ w2bB,
    const bf16_t* __restrict__ w3aB, const bf16_t* __restrict__ w3bB,
    float* __restrict__ out) {
    __shared__ __align__(16) unsigned char pool[POOL_BYTES];

    const int tid  = threadIdx.x;
    const int lane = tid & 31;
    const int wave = tid >> 5;
    const int half = lane >> 4, col = lane & 15;
    const int b = blockIdx.x >> 8;
    const int t = blockIdx.x & 255;

    // LDS pool layout (regions reused across pipeline stages; all bf16
    // matrices use the swizzled (K=row, N=col) storage; sizes = rows*64B)
    bf16_t* gTS   = (bf16_t*)(pool + 0);      // 32 rows   (2 KB)
    bf16_t* inpS  = (bf16_t*)(pool + 2048);   // 128 rows  (8 KB)
    bf16_t* qS    = (bf16_t*)(pool + 10240);  // 256 rows  (16 KB)
    bf16_t* kS    = (bf16_t*)(pool + 26624);  // 256 rows  (16 KB)
    float*  scF   = (float*)(pool + 43008);   // [32][32] f32 scores (4 KB)
    bf16_t* agg12 = qS;                       // agg for gcn1/gcn2
    bf16_t* h1S   = kS;                       // 128 rows
    bf16_t* h2S   = (bf16_t*)(pool + 34816);  // 256 rows  (16 KB)
    bf16_t* agg3  = (bf16_t*)(pool + 2048);   // 256 rows  (16 KB)
    bf16_t* h3S   = (bf16_t*)(pool + 18432);  // 256 rows  (16 KB)
    // stage-1 only (overlap q/k regions, dead until stage 2):
    bf16_t* E1    = qS;                       // 3 x 2048 elems (12 KB)
    float*  pn    = (float*)(pool + 26624);   // [3][32] normalized pos
    float*  dn    = pn + 96;                  // [3][32] normalized dif
    float*  spaL  = pn + 192;                 // spa staged (625 f32)

    if (lane == 0) {  // warm L2 for the biggest weights (global_prefetch_b8)
        __builtin_prefetch(w3aB, 0, 1);
        __builtin_prefetch(w3bB, 0, 1);
    }

    // zero pool (gTS pad rows must be 0)
    for (int i = tid; i < POOL_BYTES / 4; i += 256) ((int*)pool)[i] = 0;
    __syncthreads();

    // ---------------- stage 1a: stage spa + normalized pos/dif ------------
    for (int i = tid; i < 625; i += 256) spaL[i] = spa[i];
    if (tid < 96) {
        int c = tid >> 5, j = tid & 31;
        if (j < J) {
            int xi = ((b * 3 + c) * J + j) * TT + t;
            float xv = x[xi];
            float dv = (t > 0) ? (xv - x[xi - 1]) : 0.f;
            pn[c * 32 + j] = xv * (jbn_g[c * J + j] * RS) + jbn_b[c * J + j];
            dn[c * 32 + j] = dv * (dbn_g[c * J + j] * RS) + dbn_b[c * J + j];
        }
    }
    __syncthreads();

    // ---------------- stage 1b: embed layer-1 -> E1 (3 paths, 64x32) ------
    for (int it = 0; it < 24; ++it) {
        int idx = tid + (it << 8);
        int p = idx >> 11, r = idx & 2047;
        int j = r & 31, c = r >> 5;
        float v = 0.f;
        if (j < J) {
            if (p == 0)
                v = jw1[c*3]*pn[j] + jw1[c*3+1]*pn[32+j] + jw1[c*3+2]*pn[64+j]
                    + jb1[c];
            else if (p == 1)
                v = dw1[c*3]*dn[j] + dw1[c*3+1]*dn[32+j] + dw1[c*3+2]*dn[64+j]
                    + db1[c];
            else {
                v = sb1[c];
                for (int ci = 0; ci < J; ++ci)
                    v += sw1[c * J + ci] * spaL[ci * J + j];
            }
            v = fmaxf(v, 0.f);
        }
        E1[(p << 11) + swz_idx(c, j)] = (bf16_t)v;
    }
    __syncthreads();

    // ---------------- stage 1c: embed layer-2 via WMMA -> inpS ------------
    {   // 8 jobs == 8 waves: (mt, nt) over 64 rows x 32 cols
        int m0 = (wave >> 1) << 4, n0 = (wave & 1) << 4;
        int mt = m0 >> 4;
        v8f accP, accD, accS;
#pragma unroll
        for (int r = 0; r < 8; ++r) {
            int row = m0 + r + half * 8;
            accP[r] = jb2[row]; accD[r] = db2[row]; accS[r] = sb2[row];
        }
#pragma unroll
        for (int kk = 0; kk < 2; ++kk) {
            accP = wmma_bf16(load_a_w(jw2B, 2, mt, kk, lane),
                             load_b_swz(E1, kk * 32, n0, lane), accP);
            accD = wmma_bf16(load_a_w(dw2B, 2, mt, kk, lane),
                             load_b_swz(E1 + 2048, kk * 32, n0, lane), accD);
            accS = wmma_bf16(load_a_w(sw2B, 2, mt, kk, lane),
                             load_b_swz(E1 + 4096, kk * 32, n0, lane), accS);
        }
        v8bf pk1, pk2;
#pragma unroll
        for (int r = 0; r < 8; ++r) {
            pk1[r] = (bf16_t)(fmaxf(accP[r], 0.f) + fmaxf(accD[r], 0.f));
            pk2[r] = (bf16_t)fmaxf(accS[r], 0.f);
        }
        *(v8bf*)(inpS + swz_idx(m0 + half * 8, n0 + col)) = pk1;
        *(v8bf*)(inpS + swz_idx(64 + m0 + half * 8, n0 + col)) = pk2;
    }
    __syncthreads();

    // ---------------- stage 2: q,k = W[256,128] x inp (WMMA) --------------
    for (int job = wave; job < 64; job += 8) {
        bool isQ = job < 32;
        const bf16_t* W = isQ ? g1wB : g2wB;
        const float* bias = isQ ? g1b : g2b;
        bf16_t* dst = isQ ? qS : kS;
        int jj = job & 31;
        int m0 = (jj >> 1) << 4, n0 = (jj & 1) << 4;
        v8f acc;
#pragma unroll
        for (int r = 0; r < 8; ++r) acc[r] = bias[m0 + r + half * 8];
#pragma unroll
        for (int kk = 0; kk < 4; ++kk) {
            v16bf a = load_a_w(W, 4, m0 >> 4, kk, lane);
            v16bf bm = load_b_swz(inpS, kk * 32, n0, lane);
            acc = wmma_bf16(a, bm, acc);
        }
        store_tile_swz(dst, m0, n0, acc, lane);
    }
    __syncthreads();

    // ---------------- stage 3: scores = q^T k, softmax -> gTS -------------
    for (int job = wave; job < 4; job += 8) {
        int m0 = (job >> 1) << 4, n0 = (job & 1) << 4;
        v8f acc = {};
#pragma unroll
        for (int kk = 0; kk < 8; ++kk) {
            v16bf a = load_at_act(qS, m0, kk * 32, lane);
            v16bf bm = load_b_swz(kS, kk * 32, n0, lane);
            acc = wmma_bf16(a, bm, acc);
        }
#pragma unroll
        for (int r = 0; r < 8; ++r)
            scF[(m0 + r + half * 8) * JP + n0 + col] = acc[r];
    }
    __syncthreads();
    if (tid < J) {
        float vals[J];
        float m = -1e30f;
#pragma unroll
        for (int k2 = 0; k2 < J; ++k2) {
            vals[k2] = scF[tid * JP + k2];
            m = fmaxf(m, vals[k2]);
        }
        float s = 0.f;
#pragma unroll
        for (int k2 = 0; k2 < J; ++k2) { vals[k2] = __expf(vals[k2] - m); s += vals[k2]; }
        float inv = 1.f / s;
#pragma unroll
        for (int k2 = 0; k2 < J; ++k2)           // gT[k][j] = g[j][k]
            gTS[swz_idx(k2, tid)] = (bf16_t)(vals[k2] * inv);
    }
    __syncthreads();

    // ---------------- stages 4-6: three gcn_spa layers --------------------
    gcn_layer(128, 128, lane, wave, inpS, agg12, h1S, gTS, w1aB, w1bB,
              w1bb, bn1g, bn1b);
    gcn_layer(128, 256, lane, wave, h1S, agg12, h2S, gTS, w2aB, w2bB,
              w2bb, bn2g, bn2b);
    gcn_layer(256, 256, lane, wave, h2S, agg3, h3S, gTS, w3aB, w3bB,
              w3bb, bn3g, bn3b);

    // ---------------- stage 7: max over joints -> out [B,256,1,T] ---------
    {
        int row = tid;  // 256 channels
        float m = -1e30f;
#pragma unroll
        for (int j2 = 0; j2 < J; ++j2)
            m = fmaxf(m, (float)h3S[swz_idx(row, j2)]);
        out[(b * 256 + row) * TT + t] = m;
    }
}

extern "C" void kernel_launch(void* const* d_in, const int* in_sizes, int n_in,
                              void* d_out, int out_size, void* d_ws, size_t ws_size,
                              hipStream_t stream) {
    const float* x     = (const float*)d_in[0];
    const float* spa   = (const float*)d_in[1];
    const float* jbn_g = (const float*)d_in[2];
    const float* jbn_b = (const float*)d_in[3];
    const float* jw1   = (const float*)d_in[4];
    const float* jb1   = (const float*)d_in[5];
    const float* jw2   = (const float*)d_in[6];
    const float* jb2   = (const float*)d_in[7];
    const float* dbn_g = (const float*)d_in[8];
    const float* dbn_b = (const float*)d_in[9];
    const float* dw1   = (const float*)d_in[10];
    const float* db1   = (const float*)d_in[11];
    const float* dw2   = (const float*)d_in[12];
    const float* db2   = (const float*)d_in[13];
    const float* sw1   = (const float*)d_in[14];
    const float* sb1   = (const float*)d_in[15];
    const float* sw2   = (const float*)d_in[16];
    const float* sb2   = (const float*)d_in[17];
    const float* g1w   = (const float*)d_in[18];
    const float* g1b   = (const float*)d_in[19];
    const float* g2w   = (const float*)d_in[20];
    const float* g2b   = (const float*)d_in[21];
    const float* w1a   = (const float*)d_in[22];
    const float* w1b_  = (const float*)d_in[23];
    const float* w1bb  = (const float*)d_in[24];
    const float* bn1g  = (const float*)d_in[25];
    const float* bn1b  = (const float*)d_in[26];
    const float* w2a   = (const float*)d_in[27];
    const float* w2b_  = (const float*)d_in[28];
    const float* w2bb  = (const float*)d_in[29];
    const float* bn2g  = (const float*)d_in[30];
    const float* bn2b  = (const float*)d_in[31];
    const float* w3a   = (const float*)d_in[32];
    const float* w3b_  = (const float*)d_in[33];
    const float* w3bb  = (const float*)d_in[34];
    const float* bn3g  = (const float*)d_in[35];
    const float* bn3b  = (const float*)d_in[36];

    if (ws_size < (size_t)(307200 * sizeof(bf16_t))) return;
    bf16_t* g1wB = (bf16_t*)d_ws;
    bf16_t* g2wB = g1wB + 32768;
    bf16_t* w1aB = g2wB + 32768;
    bf16_t* w1bB = w1aB + 16384;
    bf16_t* w2aB = w1bB + 16384;
    bf16_t* w2bB = w2aB + 32768;
    bf16_t* w3aB = w2bB + 32768;
    bf16_t* w3bB = w3aB + 65536;
    bf16_t* jw2B = w3bB + 65536;
    bf16_t* dw2B = jw2B + 4096;
    bf16_t* sw2B = dw2B + 4096;

    // prologue: fp32 weights -> bf16, pre-swizzled into A-operand tile order
    swz_weight<<<128, 256, 0, stream>>>(g1w, g1wB, 256, 128);
    swz_weight<<<128, 256, 0, stream>>>(g2w, g2wB, 256, 128);
    swz_weight<<<64, 256, 0, stream>>>(w1a, w1aB, 128, 128);
    swz_weight<<<64, 256, 0, stream>>>(w1b_, w1bB, 128, 128);
    swz_weight<<<128, 256, 0, stream>>>(w2a, w2aB, 256, 128);
    swz_weight<<<128, 256, 0, stream>>>(w2b_, w2bB, 256, 128);
    swz_weight<<<256, 256, 0, stream>>>(w3a, w3aB, 256, 256);
    swz_weight<<<256, 256, 0, stream>>>(w3b_, w3bB, 256, 256);
    swz_weight<<<16, 256, 0, stream>>>(jw2, jw2B, 64, 64);
    swz_weight<<<16, 256, 0, stream>>>(dw2, dw2B, 64, 64);
    swz_weight<<<16, 256, 0, stream>>>(sw2, sw2B, 64, 64);

    sgn_fused<<<dim3(BB * TT), dim3(256), 0, stream>>>(
        x, spa, jbn_g, jbn_b, jw1, jb1, jb2, dbn_g, dbn_b, dw1, db1, db2,
        sw1, sb1, sb2, g1b, g2b, w1bb, bn1g, bn1b, w2bb, bn2g, bn2b,
        w3bb, bn3g, bn3b, jw2B, dw2B, sw2B,
        g1wB, g2wB, w1aB, w1bB, w2aB, w2bB, w3aB, w3bB,
        (float*)d_out);
}
